// Range_Fourier_Net_17162689315049
// MI455X (gfx1250) — matrix operand order repaired and checked by reading, not compiled, not run on hardware
//
#include <hip/hip_runtime.h>
#include <hip/hip_bf16.h>

// DFT-as-GEMM for MI455X (gfx1250, wave32, WMMA).
// y_r = x_r W_r - x_i W_i ; y_i = x_r W_i + x_i W_r, W symmetric (W^T == W).
// M = 65536 rows, N = K = 256. f16 fragments, f32 accumulate via
// v_wmma_f32_16x16x32_f16. Four accumulators avoid unsupported A/B NEG bits.

typedef __attribute__((ext_vector_type(16))) _Float16 v16h;
typedef __attribute__((ext_vector_type(8)))  float    v8f;

#define M_TOTAL   65536
#define NDIM      256
#define REIM_OFF  (M_TOTAL * NDIM)   // offset of imaginary plane in d_out
#define WAVES_PER_BLOCK 8

__global__ __launch_bounds__(256, 1)
void range_dft_wmma_f16(const float* __restrict__ xr,
                        const float* __restrict__ xi,
                        const float* __restrict__ wr,
                        const float* __restrict__ wi,
                        float* __restrict__ out) {
    // Staged f16 column panel of W: rows k=0..255, cols n=0..15 (per nt tile).
    __shared__ __align__(32) _Float16 lwr[NDIM * 16];
    __shared__ __align__(32) _Float16 lwi[NDIM * 16];

    const int tid  = threadIdx.x;
    const int lane = tid & 31;
    const int wave = tid >> 5;
    const int row0 = (blockIdx.x * WAVES_PER_BLOCK + wave) * 16; // 16-row M tile per wave

    // A-fragment addressing per ISA 16-bit A 16x32 layout:
    // lanes 0-15: M = lane,    K halves = {0..7, 16..23}
    // lanes 16-31: M = lane-16, K halves = {8..15, 24..31}
    const int m     = lane & 15;
    const int kbase = (lane & 16) ? 8 : 0;
    const float* xr_row = xr + (size_t)(row0 + m) * NDIM;
    const float* xi_row = xi + (size_t)(row0 + m) * NDIM;

    // Load all A fragments once (x read from HBM exactly once), cvt f32->f16.
    v16h Ar[8], Ai[8];
#pragma unroll
    for (int kt = 0; kt < 8; ++kt) {
        const float* pr = xr_row + kt * 32 + kbase;
        const float* pi = xi_row + kt * 32 + kbase;
#pragma unroll
        for (int u = 0; u < 8; ++u) {
            Ar[kt][u]     = (_Float16)pr[u];
            Ar[kt][u + 8] = (_Float16)pr[u + 16];
            Ai[kt][u]     = (_Float16)pi[u];
            Ai[kt][u + 8] = (_Float16)pi[u + 16];
        }
    }

    // C/D layout: VGPR v, lane l -> (row = v + (l>=16 ? 8 : 0), col = l&15)
    const int mbase = row0 + ((lane >> 4) << 3);

    for (int nt = 0; nt < 16; ++nt) {
        __syncthreads();  // previous panel fully consumed by all waves
        // Stage W[:, nt*16 .. nt*16+16) as f16: thread t handles row k = t.
        {
            const float* prow_r = wr + (size_t)tid * NDIM + nt * 16;
            const float* prow_i = wi + (size_t)tid * NDIM + nt * 16;
#pragma unroll
            for (int u = 0; u < 16; ++u) {
                lwr[tid * 16 + u] = (_Float16)prow_r[u];
                lwi[tid * 16 + u] = (_Float16)prow_i[u];
            }
        }
        __syncthreads();

        // P = x_r W_r, Q = x_i W_i, S = x_r W_i, T = x_i W_r
        v8f P = {}, Q = {}, S = {}, T = {};
#pragma unroll
        for (int kt = 0; kt < 8; ++kt) {
            // B layout: lane -> N col (lane&15), halves = K {kbase..kbase+7, 16+kbase..16+kbase+7}.
            v16h Br, Bi;
#pragma unroll
            for (int u = 0; u < 8; ++u) {
                const int krow = kt * 32 + kbase + u;
                Br[u]     = lwr[krow * 16 + m];
                Br[u + 8] = lwr[(krow + 16) * 16 + m];
                Bi[u]     = lwi[krow * 16 + m];
                Bi[u + 8] = lwi[(krow + 16) * 16 + m];
            }
            P = __builtin_amdgcn_wmma_f32_16x16x32_f16(false, Ar[kt], false, Br,
                                                       (short)0, P, false, false);
            Q = __builtin_amdgcn_wmma_f32_16x16x32_f16(false, Ai[kt], false, Bi,
                                                       (short)0, Q, false, false);
            S = __builtin_amdgcn_wmma_f32_16x16x32_f16(false, Ar[kt], false, Bi,
                                                       (short)0, S, false, false);
            T = __builtin_amdgcn_wmma_f32_16x16x32_f16(false, Ai[kt], false, Br,
                                                       (short)0, T, false, false);
        }

        // Combine and store: out_r = P - Q, out_i = S + T.
        const int n0 = nt * 16 + (lane & 15);
#pragma unroll
        for (int v = 0; v < 8; ++v) {
            const size_t r = (size_t)(mbase + v) * NDIM + n0;
            out[r]            = P[v] - Q[v];
            out[REIM_OFF + r] = S[v] + T[v];
        }
    }
}

extern "C" void kernel_launch(void* const* d_in, const int* in_sizes, int n_in,
                              void* d_out, int out_size, void* d_ws, size_t ws_size,
                              hipStream_t stream) {
    (void)in_sizes; (void)n_in; (void)d_ws; (void)ws_size; (void)out_size;
    const float* xr = (const float*)d_in[0];
    const float* xi = (const float*)d_in[1];
    const float* wr = (const float*)d_in[2];
    const float* wi = (const float*)d_in[3];
    float* out = (float*)d_out;

    // 65536 rows / (16 rows per wave * 8 waves per block) = 512 blocks.
    dim3 grid(M_TOTAL / (16 * WAVES_PER_BLOCK));
    dim3 block(256);
    range_dft_wmma_f16<<<grid, block, 0, stream>>>(xr, xi, wr, wi, out);
}
